// ManualMultiHeadSelfAttention_89283780149513
// MI455X (gfx1250) — matrix operand order repaired and checked by reading, not compiled, not run on hardware
//
#include <hip/hip_runtime.h>
#include <hip/hip_bf16.h>

#define BATCH  8
#define SEQ    1024
#define DMODEL 768
#define HEADS  12
#define HDIM   64

typedef __attribute__((ext_vector_type(16))) __bf16 bf16x16;
typedef __attribute__((ext_vector_type(8)))  __bf16 bf16x8;
typedef __attribute__((ext_vector_type(8)))  float  f32x8;

union ABfrag { bf16x16 v; bf16x8 h[2]; };

__device__ __forceinline__ unsigned short f2bf(float f) {
    union { float f; unsigned int u; } c; c.f = f;
    unsigned int u = c.u;
    u += 0x7fffu + ((u >> 16) & 1u);          // round to nearest even
    return (unsigned short)(u >> 16);
}

// ---------------- conversion kernels ----------------
__global__ void cvt_bf16(const float* __restrict__ s, unsigned short* __restrict__ d, int n) {
    int i = blockIdx.x * blockDim.x + threadIdx.x;
    if (i < n) d[i] = f2bf(s[i]);
}

// W stored [in, out] (row-major) -> Wt [out, in] bf16 (so B-fragment loads are contiguous)
__global__ void cvtT_bf16(const float* __restrict__ s, unsigned short* __restrict__ d) {
    int t = blockIdx.x * blockDim.x + threadIdx.x;
    if (t < DMODEL * DMODEL) {
        int o = t / DMODEL, i = t % DMODEL;
        d[t] = f2bf(s[i * DMODEL + o]);       // coalesced writes, strided reads
    }
}

// ---------------- QKV projection: one wave per 64x64 macro-tile (4x4 fragments) ----------------
__global__ __launch_bounds__(32)
void qkv_gemm(const unsigned short* __restrict__ xb,
              const unsigned short* __restrict__ Wtq,
              const unsigned short* __restrict__ Wtk,
              const unsigned short* __restrict__ Wtv,
              const float* __restrict__ bq,
              const float* __restrict__ bk,
              const float* __restrict__ bv,
              unsigned short* __restrict__ Q,   // [B,H,N,HDIM]
              unsigned short* __restrict__ K,   // [B,H,N,HDIM]
              unsigned short* __restrict__ Vt)  // [B,H,HDIM,N] (transposed)
{
    const int l = threadIdx.x, lo = l & 15, hi = l >> 4;
    const int m0 = blockIdx.x * 64;           // row macro-tile over B*N = 8192
    const int n0 = blockIdx.y * 64;           // col macro-tile over 768
    const int mode = blockIdx.z;              // 0=Q 1=K 2=V

    const unsigned short* Wt   = (mode == 0) ? Wtq : (mode == 1) ? Wtk : Wtv;
    const float*          bias = (mode == 0) ? bq  : (mode == 1) ? bk  : bv;

    const unsigned short* arow[4];
    const unsigned short* brow[4];
#pragma unroll
    for (int i = 0; i < 4; ++i) {
        arow[i] = xb + (size_t)(m0 + i * 16 + lo) * DMODEL;
        brow[i] = Wt + (size_t)(n0 + i * 16 + lo) * DMODEL;
    }

    f32x8 acc[4][4];
#pragma unroll
    for (int i = 0; i < 4; ++i)
#pragma unroll
        for (int j = 0; j < 4; ++j) acc[i][j] = (f32x8){};

#pragma unroll 2
    for (int kb = 0; kb < DMODEL; kb += 32) {
        ABfrag a[4];
        bf16x16 bw[4];
#pragma unroll
        for (int i = 0; i < 4; ++i) {
            a[i].h[0] = *(const bf16x8*)(arow[i] + kb + hi * 16);
            a[i].h[1] = *(const bf16x8*)(arow[i] + kb + hi * 16 + 8);
            bw[i]     = *(const bf16x16*)(brow[i] + kb + hi * 16);
        }
#pragma unroll
        for (int i = 0; i < 4; ++i)
#pragma unroll
            for (int j = 0; j < 4; ++j)
                acc[i][j] = __builtin_amdgcn_wmma_f32_16x16x32_bf16(
                    false, a[i].v, false, bw[j], (short)0, acc[i][j], false, false);
    }

#pragma unroll
    for (int j = 0; j < 4; ++j) {
        const int col = n0 + j * 16 + lo;
        const float bb = bias[col];
        const int h = col / HDIM, dd = col % HDIM;
#pragma unroll
        for (int i = 0; i < 4; ++i) {
#pragma unroll
            for (int r = 0; r < 8; ++r) {
                int gr = m0 + i * 16 + r + hi * 8;    // global row in [0, B*N)
                int b  = gr >> 10, n = gr & 1023;
                unsigned short val = f2bf(acc[i][j][r] + bb);
                size_t bh = (size_t)(b * HEADS + h);
                if (mode == 0)      Q [(bh * SEQ  + n ) * HDIM + dd] = val;
                else if (mode == 1) K [(bh * SEQ  + n ) * HDIM + dd] = val;
                else                Vt[(bh * HDIM + dd) * SEQ  + n ] = val;
            }
        }
    }
}

// ---------------- flash attention: one wave per (b, h, 16-query tile) ----------------
__global__ __launch_bounds__(32)
void attn_kernel(const unsigned short* __restrict__ Q,
                 const unsigned short* __restrict__ K,
                 const unsigned short* __restrict__ Vt,
                 unsigned short* __restrict__ ctx)   // [B*N, DMODEL] bf16
{
    __shared__ unsigned short pb[16 * 32];           // P staging: 16 rows x 32 keys
    const int l = threadIdx.x, lo = l & 15, hi = l >> 4;
    const int q0 = blockIdx.x * 16;
    const int h  = blockIdx.y;
    const int b  = blockIdx.z;
    const size_t bh = (size_t)(b * HEADS + h);

    const unsigned short* Qp = Q  + (bh * SEQ + q0 + lo) * HDIM;
    const unsigned short* Kp = K  + bh * SEQ  * HDIM;
    const unsigned short* Vp = Vt + bh * HDIM * SEQ;

    ABfrag aq0, aq1;                                  // Q rows, K-dim 0..31 / 32..63
    aq0.h[0] = *(const bf16x8*)(Qp + hi * 16);
    aq0.h[1] = *(const bf16x8*)(Qp + hi * 16 + 8);
    aq1.h[0] = *(const bf16x8*)(Qp + 32 + hi * 16);
    aq1.h[1] = *(const bf16x8*)(Qp + 32 + hi * 16 + 8);

    f32x8 acc0 = {}, acc1 = {}, acc2 = {}, acc3 = {};
    float rm[8], rl[8];
#pragma unroll
    for (int r = 0; r < 8; ++r) { rm[r] = -3.0e38f; rl[r] = 0.f; }
    const float scale = 0.125f;                       // 1/sqrt(64)

    for (int c = 0; c < SEQ / 32; ++c) {
        // ---- S = Q K^T for 32 keys (two 16x16 tiles) ----
        f32x8 s0 = {}, s1 = {};
        {
            const unsigned short* kr = Kp + (size_t)(c * 32 + lo) * HDIM;
            __builtin_prefetch(kr + 32 * HDIM, 0, 0); // next chunk -> global_prefetch_b8
            bf16x16 b0 = *(const bf16x16*)(kr + hi * 16);
            bf16x16 b1 = *(const bf16x16*)(kr + 32 + hi * 16);
            s0 = __builtin_amdgcn_wmma_f32_16x16x32_bf16(false, aq0.v, false, b0, (short)0, s0, false, false);
            s0 = __builtin_amdgcn_wmma_f32_16x16x32_bf16(false, aq1.v, false, b1, (short)0, s0, false, false);
        }
        {
            const unsigned short* kr = Kp + (size_t)(c * 32 + 16 + lo) * HDIM;
            bf16x16 b0 = *(const bf16x16*)(kr + hi * 16);
            bf16x16 b1 = *(const bf16x16*)(kr + 32 + hi * 16);
            s1 = __builtin_amdgcn_wmma_f32_16x16x32_bf16(false, aq0.v, false, b0, (short)0, s1, false, false);
            s1 = __builtin_amdgcn_wmma_f32_16x16x32_bf16(false, aq1.v, false, b1, (short)0, s1, false, false);
        }

        // ---- online softmax over this 32-key chunk ----
        // C layout: lane holds col=lo, VGPR r holds row (r + hi*8); a row lives in one 16-lane half
#pragma unroll
        for (int r = 0; r < 8; ++r) {
            float v0 = s0[r] * scale, v1 = s1[r] * scale;
            float tm = fmaxf(v0, v1);
            tm = fmaxf(tm, __shfl_xor(tm, 1));
            tm = fmaxf(tm, __shfl_xor(tm, 2));
            tm = fmaxf(tm, __shfl_xor(tm, 4));
            tm = fmaxf(tm, __shfl_xor(tm, 8));
            float nm    = fmaxf(rm[r], tm);
            float alpha = __expf(rm[r] - nm);
            float p0 = __expf(v0 - nm);
            float p1 = __expf(v1 - nm);
            float ts = p0 + p1;
            ts += __shfl_xor(ts, 1);
            ts += __shfl_xor(ts, 2);
            ts += __shfl_xor(ts, 4);
            ts += __shfl_xor(ts, 8);
            rl[r] = rl[r] * alpha + ts;
            rm[r] = nm;
            acc0[r] *= alpha; acc1[r] *= alpha; acc2[r] *= alpha; acc3[r] *= alpha;
            int M = r + hi * 8;
            pb[M * 32 + lo]      = f2bf(p0);
            pb[M * 32 + 16 + lo] = f2bf(p1);
        }
        __syncthreads();

        // ---- acc += P (16x32) @ V (32x64) ----
        ABfrag pa;                                    // A layout from LDS
    pa.h[0] = *(const bf16x8*)(&pb[lo * 32 + hi * 16]);
    pa.h[1] = *(const bf16x8*)(&pb[lo * 32 + hi * 16 + 8]);
        const unsigned short* vr = Vp + c * 32 + hi * 16;
        bf16x16 bv0 = *(const bf16x16*)(vr + (size_t)(0 * 16 + lo) * SEQ);
        bf16x16 bv1 = *(const bf16x16*)(vr + (size_t)(1 * 16 + lo) * SEQ);
        bf16x16 bv2 = *(const bf16x16*)(vr + (size_t)(2 * 16 + lo) * SEQ);
        bf16x16 bv3 = *(const bf16x16*)(vr + (size_t)(3 * 16 + lo) * SEQ);
        acc0 = __builtin_amdgcn_wmma_f32_16x16x32_bf16(false, pa.v, false, bv0, (short)0, acc0, false, false);
        acc1 = __builtin_amdgcn_wmma_f32_16x16x32_bf16(false, pa.v, false, bv1, (short)0, acc1, false, false);
        acc2 = __builtin_amdgcn_wmma_f32_16x16x32_bf16(false, pa.v, false, bv2, (short)0, acc2, false, false);
        acc3 = __builtin_amdgcn_wmma_f32_16x16x32_bf16(false, pa.v, false, bv3, (short)0, acc3, false, false);
        __syncthreads();
    }

    // ---- normalize and scatter into ctx [B*N, 768] at column block h*64 ----
#pragma unroll
    for (int r = 0; r < 8; ++r) {
        int gq = q0 + r + hi * 8;
        float inv = 1.0f / rl[r];
        size_t off = ((size_t)b * SEQ + gq) * DMODEL + h * HDIM + lo;
        ctx[off +  0] = f2bf(acc0[r] * inv);
        ctx[off + 16] = f2bf(acc1[r] * inv);
        ctx[off + 32] = f2bf(acc2[r] * inv);
        ctx[off + 48] = f2bf(acc3[r] * inv);
    }
}

// ---------------- output projection: one wave per 64x64 macro-tile, f32 out ----------------
__global__ __launch_bounds__(32)
void out_gemm(const unsigned short* __restrict__ ctx,
              const unsigned short* __restrict__ Wto,   // [out, in] bf16
              const float* __restrict__ bo,
              float* __restrict__ out)                  // [B*N, 768] f32
{
    const int l = threadIdx.x, lo = l & 15, hi = l >> 4;
    const int m0 = blockIdx.x * 64;
    const int n0 = blockIdx.y * 64;

    const unsigned short* arow[4];
    const unsigned short* brow[4];
#pragma unroll
    for (int i = 0; i < 4; ++i) {
        arow[i] = ctx + (size_t)(m0 + i * 16 + lo) * DMODEL;
        brow[i] = Wto + (size_t)(n0 + i * 16 + lo) * DMODEL;
    }

    f32x8 acc[4][4];
#pragma unroll
    for (int i = 0; i < 4; ++i)
#pragma unroll
        for (int j = 0; j < 4; ++j) acc[i][j] = (f32x8){};

#pragma unroll 2
    for (int kb = 0; kb < DMODEL; kb += 32) {
        ABfrag a[4];
        bf16x16 bw[4];
#pragma unroll
        for (int i = 0; i < 4; ++i) {
            a[i].h[0] = *(const bf16x8*)(arow[i] + kb + hi * 16);
            a[i].h[1] = *(const bf16x8*)(arow[i] + kb + hi * 16 + 8);
            bw[i]     = *(const bf16x16*)(brow[i] + kb + hi * 16);
        }
#pragma unroll
        for (int i = 0; i < 4; ++i)
#pragma unroll
            for (int j = 0; j < 4; ++j)
                acc[i][j] = __builtin_amdgcn_wmma_f32_16x16x32_bf16(
                    false, a[i].v, false, bw[j], (short)0, acc[i][j], false, false);
    }

#pragma unroll
    for (int j = 0; j < 4; ++j) {
        const int col = n0 + j * 16 + lo;
        const float bb = bo[col];
#pragma unroll
        for (int i = 0; i < 4; ++i) {
#pragma unroll
            for (int r = 0; r < 8; ++r) {
                int gr = m0 + i * 16 + r + hi * 8;
                out[(size_t)gr * DMODEL + col] = acc[i][j][r] + bb;
            }
        }
    }
}

// ---------------- host side ----------------
extern "C" void kernel_launch(void* const* d_in, const int* in_sizes, int n_in,
                              void* d_out, int out_size, void* d_ws, size_t ws_size,
                              hipStream_t stream) {
    (void)in_sizes; (void)n_in; (void)out_size; (void)ws_size;
    const float* x  = (const float*)d_in[0];
    const float* Wq = (const float*)d_in[1];
    const float* bq = (const float*)d_in[2];
    const float* Wk = (const float*)d_in[3];
    const float* bk = (const float*)d_in[4];
    const float* Wv = (const float*)d_in[5];
    const float* bv = (const float*)d_in[6];
    const float* Wo = (const float*)d_in[7];
    const float* bo = (const float*)d_in[8];
    float* out = (float*)d_out;

    char* p = (char*)d_ws;
    auto carve = [&](size_t elems) {
        unsigned short* r = (unsigned short*)p;
        p += ((elems * sizeof(unsigned short) + 255) / 256) * 256;
        return r;
    };
    const size_t nX   = (size_t)BATCH * SEQ * DMODEL;       // 6.29M elems
    const size_t nW   = (size_t)DMODEL * DMODEL;            // 590K elems
    const size_t nQKV = (size_t)BATCH * HEADS * SEQ * HDIM; // 6.29M elems

    unsigned short* xb   = carve(nX);
    unsigned short* Wtq  = carve(nW);
    unsigned short* Wtk  = carve(nW);
    unsigned short* Wtv  = carve(nW);
    unsigned short* Wto  = carve(nW);
    unsigned short* Qb   = carve(nQKV);
    unsigned short* Kb   = carve(nQKV);
    unsigned short* Vtb  = carve(nQKV);
    unsigned short* ctxb = carve(nX);

    cvt_bf16 <<<(int)((nX + 255) / 256), 256, 0, stream>>>(x, xb, (int)nX);
    cvtT_bf16<<<(int)((nW + 255) / 256), 256, 0, stream>>>(Wq, Wtq);
    cvtT_bf16<<<(int)((nW + 255) / 256), 256, 0, stream>>>(Wk, Wtk);
    cvtT_bf16<<<(int)((nW + 255) / 256), 256, 0, stream>>>(Wv, Wtv);
    cvtT_bf16<<<(int)((nW + 255) / 256), 256, 0, stream>>>(Wo, Wto);

    qkv_gemm<<<dim3(BATCH * SEQ / 64, DMODEL / 64, 3), 32, 0, stream>>>(
        xb, Wtq, Wtk, Wtv, bq, bk, bv, Qb, Kb, Vtb);

    attn_kernel<<<dim3(SEQ / 16, HEADS, BATCH), 32, 0, stream>>>(Qb, Kb, Vtb, ctxb);

    out_gemm<<<dim3(BATCH * SEQ / 64, DMODEL / 64), 32, 0, stream>>>(ctxb, Wto, bo, out);
}
